// HierarchicalPolicyHead_18708877542138
// MI455X (gfx1250) — matrix operand order repaired and checked by reading, not compiled, not run on hardware
//
#include <hip/hip_runtime.h>
#include <hip/hip_bf16.h>
#include <math.h>

typedef __attribute__((ext_vector_type(16))) __bf16 bf16x16;
typedef __attribute__((ext_vector_type(8)))  __bf16 bf16x8;
typedef __attribute__((ext_vector_type(8)))  float  f32x8;

#define NEGV (-1000000000.0f)
#define NBATCH 4096
#define SP 81
#define MROWS (NBATCH * SP)          // 331776, divisible by 16
#define NACT 2026                    // 1 + 81 + 1944

// ---------------- wave32 reductions ----------------
__device__ __forceinline__ float wredmax(float v) {
#pragma unroll
  for (int o = 16; o; o >>= 1) v = fmaxf(v, __shfl_xor(v, o, 32));
  return v;
}
__device__ __forceinline__ float wredsum(float v) {
#pragma unroll
  for (int o = 16; o; o >>= 1) v += __shfl_xor(v, o, 32);
  return v;
}
__device__ __forceinline__ int wredor(int v) {
#pragma unroll
  for (int o = 16; o; o >>= 1) v |= __shfl_xor(v, o, 32);
  return v;
}

// ---------------- Kernel 1: x [b][c][s] f32 -> A row-major bf16 [b*81+s][c] ----------------
__global__ __launch_bounds__(256) void pack_x_kernel(const float* __restrict__ x,
                                                     __bf16* __restrict__ A) {
  const int b = blockIdx.x;
  const int c0 = blockIdx.y * 32;
  __shared__ __bf16 tile[32 * 84];
  const float* xs = x + (size_t)b * (256 * SP);
  for (int idx = threadIdx.x; idx < 32 * SP; idx += 256) {
    int c = idx / SP, s = idx % SP;                 // coalesced along s
    tile[c * 84 + s] = (__bf16)xs[(c0 + c) * SP + s];
  }
  __syncthreads();
  __bf16* Ab = A + (size_t)b * SP * 256 + c0;
  for (int idx = threadIdx.x; idx < SP * 32; idx += 256) {
    int s = idx / 32, c = idx % 32;                 // coalesced along c
    Ab[(size_t)s * 256 + c] = tile[c * 84 + s];
  }
}

// ---------------- Kernel 2: conv_w f32 -> bf16 pre-packed B-operand tiles ----------------
// Wp chunk layout: chunk c = (kt*16 + nt)*32 + lane ; 16 contiguous bf16 per chunk:
// col n = lane%16, K = kt*32 + (lane/16)*16 + j (j = 0..15)
__global__ __launch_bounds__(256) void pack_w_kernel(const float* __restrict__ W,
                                                     __bf16* __restrict__ Wp) {
  int c = blockIdx.x * 256 + threadIdx.x;           // 4096 chunks total
  int tile = c >> 5, lane = c & 31;
  int kt = tile >> 4, nt = tile & 15;
  int o = nt * 16 + (lane & 15);
  int k0 = kt * 32 + (lane >> 4) * 16;
  const float* src = W + o * 256 + k0;
  __bf16* dst = Wp + (size_t)c * 16;
#pragma unroll
  for (int j = 0; j < 16; ++j) dst[j] = (__bf16)src[j];
}

// ---------------- Kernel 3: zero p_sum accumulator ----------------
__global__ __launch_bounds__(256) void zero_kernel(float* __restrict__ p) {
  p[(size_t)blockIdx.x * 256 + threadIdx.x] = 0.0f;
}

// ---------------- Kernel 4: fused GEMM1 (bn+relu) + GEMM2 + p_sum ----------------
// 256 threads = 8 waves in 4 pairs. Pair p owns one 16-row M tile; wave half h
// computes output columns [h*128, h*128+128) (8 WMMA tiles = 64 acc VGPRs).
__global__ __launch_bounds__(256) void fused_kernel(
    const __bf16* __restrict__ A, const __bf16* __restrict__ Wp,
    const float* __restrict__ gma, const float* __restrict__ bta,
    const float* __restrict__ mea, const float* __restrict__ var,
    const float* __restrict__ patch_w, const float* __restrict__ patch_b,
    const float* __restrict__ pos_w, const float* __restrict__ pos_b,
    float* __restrict__ Lraw, float* __restrict__ psum) {
  __shared__ __bf16 pool[4 * 4096];   // 32 KB: 4 shared P tiles; first 8192 = W staging
  __shared__ __bf16 Wl2[16 * 512];    // 16 KB: GEMM2 weights, B-operand tiles
  __bf16* Wst = pool;                 // alias: dead before P tiles are written

  const int wave = threadIdx.x >> 5, lane = threadIdx.x & 31;
  const int pair = wave >> 1, h = wave & 1;
  const int mt = blockIdx.x * 4 + pair;
  const int mrow = mt * 16 + (lane & 15);
  const int hi = lane >> 4;

  // Stage GEMM2 weights once (f32 -> bf16, B-operand layout); visible after 1st barrier.
#pragma unroll
  for (int i = 0; i < 2; ++i) {
    int c = threadIdx.x * 2 + i;                    // 512 chunks
    int tile = c >> 5, l2 = c & 31;
    int kt = tile >> 1, nt = tile & 1;
    int o = nt * 16 + (l2 & 15);
    int k0 = kt * 32 + (l2 >> 4) * 16;
    __bf16* dst = &Wl2[(size_t)c * 16];
#pragma unroll
    for (int j = 0; j < 16; ++j) {
      int k = k0 + j;
      float v = 0.0f;
      if (o == 0) v = patch_w[k];
      else if (o < 25) v = pos_w[(o - 1) * 256 + k];
      dst[j] = (__bf16)v;
    }
  }

  // ---- main GEMM: 8 col tiles per wave, W staged per K-chunk ----
  const __bf16* aptr = A + (size_t)mrow * 256 + hi * 16;
  bf16x8 a0 = *(const bf16x8*)aptr;                 // prefetch kt=0
  bf16x8 a1 = *(const bf16x8*)(aptr + 8);
  f32x8 acc[8] = {};
  for (int kt = 0; kt < 8; ++kt) {
    __syncthreads();                                // previous LDS reads done
    // stage 16 KB of pre-packed bf16 weights for this kt (pure copy, full N=256)
#pragma unroll
    for (int i = 0; i < 2; ++i) {
      int cc = threadIdx.x * 2 + i;                 // 512 chunks of 16 bf16
      *(bf16x16*)&Wst[(size_t)cc * 16] = *(const bf16x16*)(Wp + ((size_t)kt * 512 + cc) * 16);
    }
    __syncthreads();                                // staged data visible
    bf16x16 af;
#pragma unroll
    for (int j = 0; j < 8; ++j) { af[j] = a0[j]; af[8 + j] = a1[j]; }
    if (kt < 7) {                                   // prefetch next A frag during WMMAs
      a0 = *(const bf16x8*)(aptr + (kt + 1) * 32);
      a1 = *(const bf16x8*)(aptr + (kt + 1) * 32 + 8);
    }
#pragma unroll
    for (int nt = 0; nt < 8; ++nt) {
      int nt_g = h * 8 + nt;
      bf16x16 bfrag = *(const bf16x16*)&Wst[((size_t)nt_g * 32 + lane) * 16];
      acc[nt] = __builtin_amdgcn_wmma_f32_16x16x32_bf16(
          false, af, false, bfrag, (short)0, acc[nt], false, false);
    }
  }

  // ---- epilogue 1: BN + ReLU in registers ----
  const int rowbase = mt * 16 + hi * 8;
  const int col_lo = lane & 15;
#pragma unroll
  for (int nt = 0; nt < 8; ++nt) {
    int o = h * 128 + nt * 16 + col_lo;
    float sc = gma[o] * rsqrtf(var[o] + 1e-5f);
    float bs = bta[o] - mea[o] * sc;
#pragma unroll
    for (int r = 0; r < 8; ++r) acc[nt][r] = fmaxf(acc[nt][r] * sc + bs, 0.0f);
  }

  __syncthreads();                                  // all waves done with Wst before overwrite
  // ---- write own half of the pair's P tile to LDS (D-layout -> row-major) ----
  __bf16* Pt = pool + (size_t)pair * 4096;          // 16 rows x 256 cols bf16
#pragma unroll
  for (int nt = 0; nt < 8; ++nt)
#pragma unroll
    for (int r = 0; r < 8; ++r)
      Pt[(size_t)(hi * 8 + r) * 256 + h * 128 + nt * 16 + col_lo] = (__bf16)acc[nt][r];

  // ---- p_sum contributions (rows of a tile span at most 2 batches) ----
  {
    int b0 = rowbase / SP;
    int nb = (b0 + 1) * SP;                         // first row of next batch
#pragma unroll
    for (int nt = 0; nt < 8; ++nt) {
      float s0 = 0.0f, s1 = 0.0f;
#pragma unroll
      for (int r = 0; r < 8; ++r) {
        if (rowbase + r < nb) s0 += acc[nt][r]; else s1 += acc[nt][r];
      }
      int o = h * 128 + nt * 16 + col_lo;
      atomicAdd(&psum[(size_t)b0 * 256 + o], s0);
      if (rowbase + 7 >= nb) atomicAdd(&psum[(size_t)(b0 + 1) * 256 + o], s1);
    }
  }

  __syncthreads();                                  // pair's P tile complete (both halves)

  // ---- fused GEMM2: shared P tile x [patch_w; pos_w]^T ; wave handles col group h ----
  f32x8 acc2 = {};
#pragma unroll
  for (int kt = 0; kt < 8; ++kt) {
    const __bf16* ap = Pt + (size_t)(lane & 15) * 256 + kt * 32 + hi * 16;
    bf16x8 b0v = *(const bf16x8*)ap;
    bf16x8 b1v = *(const bf16x8*)(ap + 8);
    bf16x16 af;
#pragma unroll
    for (int j = 0; j < 8; ++j) { af[j] = b0v[j]; af[8 + j] = b1v[j]; }
    bf16x16 bfrag = *(const bf16x16*)&Wl2[((size_t)(kt * 2 + h) * 32 + lane) * 16];
    acc2 = __builtin_amdgcn_wmma_f32_16x16x32_bf16(
        false, af, false, bfrag, (short)0, acc2, false, false);
  }
  {
    int col = h * 16 + col_lo;
    float bias = (col == 0) ? patch_b[0] : ((col < 25) ? pos_b[col - 1] : 0.0f);
#pragma unroll
    for (int r = 0; r < 8; ++r)
      Lraw[(size_t)(rowbase + r) * 32 + col] = acc2[r] + bias;
  }
}

// ---------------- Kernel 5: type/slot/orient GEMVs (p_mean = p_sum/81) ----------------
__global__ __launch_bounds__(256) void heads_kernel(
    const float* __restrict__ psum,
    const float* __restrict__ type_w, const float* __restrict__ type_b,
    const float* __restrict__ slot_w, const float* __restrict__ slot_b,
    const float* __restrict__ orient_w, const float* __restrict__ orient_b,
    float* __restrict__ headsB) {
  const int b = blockIdx.x;
  __shared__ float pm[256];
  pm[threadIdx.x] = psum[(size_t)b * 256 + threadIdx.x] * (1.0f / 81.0f);
  __syncthreads();
  const int lane = threadIdx.x & 31, wave = threadIdx.x >> 5;
  for (int r = wave; r < 30; r += 8) {
    const float* wr; float bias;
    if (r < 3)      { wr = type_w + r * 256;         bias = type_b[r]; }
    else if (r < 6) { wr = slot_w + (r - 3) * 256;   bias = slot_b[r - 3]; }
    else            { wr = orient_w + (r - 6) * 256; bias = orient_b[r - 6]; }
    float p = 0.0f;
    for (int k = lane; k < 256; k += 32) p += wr[k] * pm[k];
    p = wredsum(p);
    if (lane == 0) headsB[(size_t)b * 32 + r] = p + bias;
  }
}

// ---------------- Kernel 6: masked log-softmaxes + joint assembly ----------------
__device__ __forceinline__ void lsm_small(const float* x, const int* mk, int n, float* lp) {
  float mx = -3.0e38f;
  for (int i = 0; i < n; ++i) mx = fmaxf(mx, mk[i] ? x[i] : NEGV);
  float s = 0.0f;
  for (int i = 0; i < n; ++i) s += __expf((mk[i] ? x[i] : NEGV) - mx);
  float l = __logf(s);
  for (int i = 0; i < n; ++i) lp[i] = fmaxf((mk[i] ? x[i] : NEGV) - mx - l, -100.0f);
}

__global__ __launch_bounds__(256) void final_kernel(const int* __restrict__ amask,
                                                    const float* __restrict__ Lraw,
                                                    const float* __restrict__ headsB,
                                                    float* __restrict__ out) {
  const int b = blockIdx.x;
  const int lane = threadIdx.x & 31, wave = threadIdx.x >> 5;
  __shared__ float sMax[25], sLse[25];
  __shared__ int sAny[25];
  __shared__ float sTypeLp[3], sSlotLp[3], sOrientLp[24];
  const int* am = amask + (size_t)b * NACT;
  const float* lr = Lraw + (size_t)b * SP * 32;
  for (int g = wave; g < 25; g += 8) {
    float mx = -3.0e38f, sum = 0.0f; int any = 0;
    float lg[3];
#pragma unroll
    for (int e = 0; e < 3; ++e) {
      int s = lane + 32 * e;
      float v = -3.0e38f;
      if (s < SP) {
        int m = (g == 24) ? am[1 + s] : am[82 + g * SP + s];
        float lo = (g == 24) ? lr[s * 32] : lr[s * 32 + 1 + g];
        v = (m > 0) ? lo : NEGV;
        any |= (m > 0) ? 1 : 0;
      }
      lg[e] = v;
      mx = fmaxf(mx, v);
    }
    mx = wredmax(mx);
#pragma unroll
    for (int e = 0; e < 3; ++e) {
      int s = lane + 32 * e;
      if (s < SP) sum += __expf(lg[e] - mx);
    }
    sum = wredsum(sum);
    any = wredor(any);
    if (lane == 0) { sMax[g] = mx; sLse[g] = __logf(sum); sAny[g] = any; }
  }
  __syncthreads();
  if (threadIdx.x == 0) {
    int buyAny = 0, slotAny[3] = {0, 0, 0};
    for (int so = 0; so < 24; ++so) { buyAny |= sAny[so]; slotAny[so >> 3] |= sAny[so]; }
    const float* hb = headsB + (size_t)b * 32;
    float tv[3] = {hb[0], hb[1], hb[2]};
    int tm[3] = {am[0] > 0 ? 1 : 0, sAny[24], buyAny};
    lsm_small(tv, tm, 3, sTypeLp);
    float sv[3] = {hb[3], hb[4], hb[5]};
    lsm_small(sv, slotAny, 3, sSlotLp);
    for (int sl = 0; sl < 3; ++sl) {
      float ov[8]; int om[8];
      for (int o = 0; o < 8; ++o) { ov[o] = hb[6 + sl * 8 + o]; om[o] = sAny[sl * 8 + o]; }
      lsm_small(ov, om, 8, &sOrientLp[sl * 8]);
    }
    out[(size_t)b * NACT] = sTypeLp[0];
  }
  __syncthreads();
  for (int g = wave; g < 25; g += 8) {
    float mx = sMax[g], lse = sLse[g];
#pragma unroll
    for (int e = 0; e < 3; ++e) {
      int s = lane + 32 * e;
      if (s < SP) {
        int m; float lo;
        if (g == 24) { m = am[1 + s]; lo = lr[s * 32]; }
        else         { m = am[82 + g * SP + s]; lo = lr[s * 32 + 1 + g]; }
        float lp = fmaxf(((m > 0) ? lo : NEGV) - mx - lse, -100.0f);
        if (g == 24)
          out[(size_t)b * NACT + 1 + s] = sTypeLp[1] + lp;
        else
          out[(size_t)b * NACT + 82 + g * SP + s] =
              sTypeLp[2] + sSlotLp[g >> 3] + sOrientLp[g] + lp;
      }
    }
  }
}

extern "C" void kernel_launch(void* const* d_in, const int* in_sizes, int n_in,
                              void* d_out, int out_size, void* d_ws, size_t ws_size,
                              hipStream_t stream) {
  const float* x        = (const float*)d_in[0];
  const int*   amask    = (const int*)d_in[1];
  const float* conv_w   = (const float*)d_in[2];
  const float* bn_gamma = (const float*)d_in[3];
  const float* bn_beta  = (const float*)d_in[4];
  const float* bn_mean  = (const float*)d_in[5];
  const float* bn_var   = (const float*)d_in[6];
  const float* type_w   = (const float*)d_in[7];
  const float* type_b   = (const float*)d_in[8];
  const float* patch_w  = (const float*)d_in[9];
  const float* patch_b  = (const float*)d_in[10];
  const float* slot_w   = (const float*)d_in[11];
  const float* slot_b   = (const float*)d_in[12];
  const float* orient_w = (const float*)d_in[13];
  const float* orient_b = (const float*)d_in[14];
  const float* pos_w    = (const float*)d_in[15];
  const float* pos_b    = (const float*)d_in[16];
  float* out = (float*)d_out;

  char* ws = (char*)d_ws;
  const size_t abytes    = (size_t)MROWS * 256 * 2;            // bf16 A: 169,869,312
  const size_t lraw_off  = abytes;                             // f32 Lraw: 42,467,328
  const size_t psum_off  = lraw_off + (size_t)MROWS * 32 * 4;  // f32 psum: 4 MB
  const size_t heads_off = psum_off + (size_t)NBATCH * 256 * 4;
  const size_t wp_off    = heads_off + (size_t)NBATCH * 32 * 4;
  __bf16* A      = (__bf16*)ws;
  float*  Lraw   = (float*)(ws + lraw_off);
  float*  psum   = (float*)(ws + psum_off);
  float*  headsB = (float*)(ws + heads_off);
  __bf16* Wp     = (__bf16*)(ws + wp_off);                     // 128 KB packed bf16 weights

  pack_x_kernel<<<dim3(NBATCH, 8), 256, 0, stream>>>(x, A);
  pack_w_kernel<<<dim3(16), 256, 0, stream>>>(conv_w, Wp);
  zero_kernel<<<dim3(NBATCH), 256, 0, stream>>>(psum);
  fused_kernel<<<dim3(MROWS / 16 / 4), 256, 0, stream>>>(
      A, Wp, bn_gamma, bn_beta, bn_mean, bn_var,
      patch_w, patch_b, pos_w, pos_b, Lraw, psum);
  heads_kernel<<<dim3(NBATCH), 256, 0, stream>>>(psum, type_w, type_b, slot_w, slot_b,
                                                 orient_w, orient_b, headsB);
  final_kernel<<<dim3(NBATCH), 256, 0, stream>>>(amask, Lraw, headsB, out);
}
